// CNNxLSTMClassifier_76982993814114
// MI455X (gfx1250) — matrix operand order repaired and checked by reading, not compiled, not run on hardware
//
#include <hip/hip_runtime.h>
#include <hip/hip_bf16.h>
#include <cmath>

// ---------------------------------------------------------------------------
// Types for CDNA5 WMMA (wave32, gfx1250)
// ---------------------------------------------------------------------------
typedef __bf16 bh;
typedef __attribute__((ext_vector_type(16))) __bf16 v16bf;
typedef __attribute__((ext_vector_type(8)))  __bf16 v8bf;
typedef __attribute__((ext_vector_type(8)))  float  v8f;
typedef __attribute__((ext_vector_type(4)))  int    v4i;
// Address-space-qualified int4 pointers for the async global->LDS builtin
// (param types per hipcc diagnostic: v4i addrspace(1)* src, v4i addrspace(3)* dst).
typedef __attribute__((address_space(1))) v4i* gas_v4i_p;
typedef __attribute__((address_space(3))) v4i* las_v4i_p;

// Async global->LDS (CDNA5): use if this toolchain declares the builtins.
#if defined(__has_builtin)
#if __has_builtin(__builtin_amdgcn_global_load_async_to_lds_b128) && \
    __has_builtin(__builtin_amdgcn_s_wait_asynccnt)
#define HAVE_ASYNC_LDS 1
#endif
#endif
#ifndef HAVE_ASYNC_LDS
#define HAVE_ASYNC_LDS 0
#endif

__device__ __forceinline__ v8f wmma_bf16(v16bf a, v16bf b, v8f c) {
  // D = A(16x32 bf16) * B(32x16 bf16) + C(16x16 f32)
  return __builtin_amdgcn_wmma_f32_16x16x32_bf16(
      /*neg_a=*/false, a, /*neg_b=*/false, b,
      /*c_mod=*/(short)0, c, /*reuse_a=*/false, /*reuse_b=*/false);
}

// A-matrix fragment (16x32, this lane = row M): per CDNA5 ISA layout
// lanes 0-15 hold K = 0..7 (v0-3) and 16..23 (v4-7); lanes 16-31 hold 8..15, 24..31.
__device__ __forceinline__ v16bf fragA(const bh* __restrict__ row, int k0, int hf) {
  v8bf a0 = *(const v8bf*)(row + k0 + hf * 8);
  v8bf a1 = *(const v8bf*)(row + k0 + 16 + hf * 8);
  v16bf a;
#pragma unroll
  for (int j = 0; j < 8; ++j) { a[j] = a0[j]; a[8 + j] = a1[j]; }
  return a;
}

// B-matrix fragment (32x16, this lane = column N): lanes 0-15 hold K=0..7, 16..23;
// lanes 16-31 hold K=8..15, 24..31. `row` points to weight[n][k], contiguous in k.
__device__ __forceinline__ v16bf fragB(const bh* __restrict__ row, int k0, int hf) {
  v8bf b0 = *(const v8bf*)(row + k0 + hf * 8);
  v8bf b1 = *(const v8bf*)(row + k0 + 16 + hf * 8);
  v16bf b;
#pragma unroll
  for (int j = 0; j < 8; ++j) { b[j] = b0[j]; b[8 + j] = b1[j]; }
  return b;
}

__device__ __forceinline__ float sigm(float x) { return 1.0f / (1.0f + expf(-x)); }

// ---------------------------------------------------------------------------
// Small utility kernels
// ---------------------------------------------------------------------------
__global__ __launch_bounds__(256) void cvt_bf16_k(const float* __restrict__ s,
                                                  bh* __restrict__ d, int n) {
  int i = blockIdx.x * 256 + threadIdx.x;
  if (i < n) d[i] = (bh)s[i];
}

__global__ __launch_bounds__(256) void fold_bn_k(const float* __restrict__ g,
                                                 const float* __restrict__ b,
                                                 const float* __restrict__ m,
                                                 const float* __restrict__ v,
                                                 const float* __restrict__ cb,
                                                 float* __restrict__ sc,
                                                 float* __restrict__ bi, int C) {
  int i = blockIdx.x * 256 + threadIdx.x;
  if (i >= C) return;
  float s = g[i] * rsqrtf(v[i] + 1e-5f);
  sc[i] = s;
  bi[i] = (cb[i] - m[i]) * s + b[i];
}

// ---------------------------------------------------------------------------
// Implicit-GEMM 3x3 conv + folded BN (+optional ReLU) via WMMA.
//
// One block = one strip of SW=NPW*16 output pixels (single image row) x ALL
// output channels. A 16B-aligned 3x(SW+16)xCin halo window is staged into LDS:
//  - interior blocks: async global->LDS b128 chunks (ASYNCcnt path), every
//    chunk maps to a valid aligned contiguous global segment;
//  - border blocks: scalar zero-filled fallback.
// Either way the WMMA inner loop's im2col gather is guard-free LDS reads.
// The 8 waves form (NPW pixel-subtile) x (NPC cout-subtile); each wave carries
// MT accumulators so one B-fragment gather feeds MT WMMAs.
// ---------------------------------------------------------------------------
template <int MT, bool RELU>
__global__ __launch_bounds__(256) void conv_tile_wmma(
    const bh* __restrict__ in,     // [B][Cin][H][W] bf16
    const bh* __restrict__ wt,     // [Cout][Cin*9] bf16 row-major
    const float* __restrict__ scale, const float* __restrict__ bias,
    bh* __restrict__ out,          // [B][Cout][H][W] bf16
    int Cin, int H, int Wd, int Cout, int SW, int NPW, int NPC) {
  __shared__ bh patch[24576];      // up to Cin=256: 256*3*32 elems (48 KB)

  const int K = Cin * 9;
  const int SWP = SW + 16;         // aligned halo window width (elements)
  const int tid = threadIdx.x;
  const int wave = tid >> 5, lane = tid & 31;
  const int hf = lane >> 4, l15 = lane & 15;

  // block -> (b, h, strip)
  const int nws = Wd / SW;
  const int ws = blockIdx.x % nws;
  const int h  = (blockIdx.x / nws) % H;
  const int b  = blockIdx.x / (nws * H);
  const int w0 = ws * SW;
  const int wA = w0 - 8;           // window start; (wA*2) is 16B aligned

  // ---- stage halo window into LDS ----
  const bool interior =
      (h >= 1) && (h + 1 < H) && (wA >= 0) && (wA + SWP <= Wd);
#if HAVE_ASYNC_LDS
  if (interior) {
    const int cpr = SWP >> 3;                  // 16B chunks per row
    const int nchunks = Cin * 3 * cpr;
    for (int c = tid; c < nchunks; c += 256) {
      int xc = c % cpr;
      int tt = c / cpr;
      int r = tt % 3;
      int ci = tt / 3;
      const bh* g =
          in + (((size_t)b * Cin + ci) * H + (h + r - 1)) * Wd + wA + xc * 8;
      __builtin_amdgcn_global_load_async_to_lds_b128(
          (gas_v4i_p)g, (las_v4i_p)(patch + (size_t)c * 8), 0, 0);
    }
    __builtin_amdgcn_s_wait_asynccnt(0);
  } else
#endif
  {
    const int total = Cin * 3 * SWP;
    for (int idx = tid; idx < total; idx += 256) {
      int x = idx % SWP;
      int tt = idx / SWP;
      int r = tt % 3;
      int ci = tt / 3;
      int ih = h + r - 1;
      int iw = wA + x;
      bh v = (bh)0.0f;
      if (ih >= 0 && ih < H && iw >= 0 && iw < Wd)
        v = in[(((size_t)b * Cin + ci) * H + ih) * Wd + iw];
      patch[idx] = v;
    }
  }
  __syncthreads();

  const int pw = wave % NPW;              // pixel subtile
  const int pc = wave / NPW;              // cout subtile base
  const int xbase = pw * 16 + l15 + 7;    // lane column in window (s=0 -> iw=w-1)

  const bh* wrow[MT];
#pragma unroll
  for (int mt = 0; mt < MT; ++mt) {
    int mo = pc + mt * NPC;
    wrow[mt] = wt + (size_t)(mo * 16 + l15) * K;
    __builtin_prefetch(wrow[mt], 0, 0);
  }

  v8f acc[MT];
#pragma unroll
  for (int mt = 0; mt < MT; ++mt)
#pragma unroll
    for (int r = 0; r < 8; ++r) acc[mt][r] = 0.0f;

  for (int k0 = 0; k0 < K; k0 += 32) {
    const bool full = (k0 + 32 <= K);
    // ---- B fragment: guard-free LDS im2col gather (shared across MT) ----
    v16bf bf;
#pragma unroll
    for (int j = 0; j < 16; ++j) {
      int k = k0 + ((j < 8) ? (hf * 8 + j) : (16 + hf * 8 + (j - 8)));
      bh val = (bh)0.0f;
      if (full || k < K) {
        int ci = k / 9; int t = k - ci * 9; int r = t / 3; int s = t - r * 3;
        val = patch[(ci * 3 + r) * SWP + xbase + s];
      }
      bf[j] = val;
    }
    // ---- A fragments + WMMA per cout subtile ----
#pragma unroll
    for (int mt = 0; mt < MT; ++mt) {
      v16bf a;
      if (full) {
        a = fragA(wrow[mt], k0, hf);
      } else {
#pragma unroll
        for (int j = 0; j < 16; ++j) {
          int k = k0 + ((j < 8) ? (hf * 8 + j) : (16 + hf * 8 + (j - 8)));
          a[j] = (k < K) ? wrow[mt][k] : (bh)0.0f;
        }
      }
      acc[mt] = wmma_bf16(a, bf, acc[mt]);
    }
  }

  // ---- epilogue: D layout -> M = r + hf*8, N = l15 ----
  const int w = w0 + pw * 16 + l15;
#pragma unroll
  for (int mt = 0; mt < MT; ++mt) {
    int mo = pc + mt * NPC;
#pragma unroll
    for (int r = 0; r < 8; ++r) {
      int co = mo * 16 + r + hf * 8;
      float v = acc[mt][r] * scale[co] + bias[co];
      if (RELU) v = v > 0.0f ? v : 0.0f;
      out[(((size_t)b * Cout + co) * H + h) * Wd + w] = (bh)v;
    }
  }
}

// ---------------------------------------------------------------------------
// Fused: 1x1 skip conv + residual add + ReLU + 2x2 maxpool.
// ---------------------------------------------------------------------------
__global__ __launch_bounds__(256) void skip_pool_k(
    const bh* __restrict__ c2,    // conv2+bn, [B][Co][H][W]
    const bh* __restrict__ xin,   // stage input, [B][Ci][H][W]
    const bh* __restrict__ skw,   // [Co][Ci]
    const float* __restrict__ skb,
    bh* __restrict__ out,         // [B][Co][H/2][W/2]
    int Bn, int Ci, int Co, int H, int Wd) {
  const int H2 = H >> 1, W2 = Wd >> 1;
  const int tot = Bn * Co * H2 * W2;
  int idx = blockIdx.x * 256 + threadIdx.x;
  if (idx >= tot) return;
  int w2 = idx % W2; int t = idx / W2;
  int h2 = t % H2;   t /= H2;
  int co = t % Co;   int b = t / Co;
  float mx = -1e30f;
#pragma unroll
  for (int dy = 0; dy < 2; ++dy)
#pragma unroll
    for (int dx = 0; dx < 2; ++dx) {
      int h = h2 * 2 + dy, w = w2 * 2 + dx;
      float s = skb[co];
      for (int ci = 0; ci < Ci; ++ci)
        s += (float)xin[(((size_t)b * Ci + ci) * H + h) * Wd + w] *
             (float)skw[(size_t)co * Ci + ci];
      float v = (float)c2[(((size_t)b * Co + co) * H + h) * Wd + w] + s;
      v = v > 0.0f ? v : 0.0f;
      mx = v > mx ? v : mx;
    }
  out[(((size_t)b * Co + co) * H2 + h2) * W2 + w2] = (bh)mx;
}

// ---------------------------------------------------------------------------
// LayerNorm over transposed features: seq[b*32+w][c*8+h] = feat[b][c][h][w].
// ---------------------------------------------------------------------------
__global__ __launch_bounds__(256) void ln_feat_k(
    const bh* __restrict__ feat,  // [16][256][8][32]
    const float* __restrict__ g, const float* __restrict__ bt,
    bh* __restrict__ out) {       // [512][2048]
  const int row = blockIdx.x;
  const int b = row >> 5, w = row & 31;
  __shared__ float red[256];
  float vals[8];
  float s = 0.0f;
#pragma unroll
  for (int i = 0; i < 8; ++i) {
    int j = threadIdx.x * 8 + i;
    int c = j >> 3, h = j & 7;
    float v = (float)feat[(((size_t)b * 256 + c) * 8 + h) * 32 + w];
    vals[i] = v; s += v;
  }
  red[threadIdx.x] = s; __syncthreads();
  for (int st = 128; st > 0; st >>= 1) {
    if (threadIdx.x < st) red[threadIdx.x] += red[threadIdx.x + st];
    __syncthreads();
  }
  float mean = red[0] * (1.0f / 2048.0f);
  __syncthreads();
  float q = 0.0f;
#pragma unroll
  for (int i = 0; i < 8; ++i) { float d = vals[i] - mean; q += d * d; }
  red[threadIdx.x] = q; __syncthreads();
  for (int st = 128; st > 0; st >>= 1) {
    if (threadIdx.x < st) red[threadIdx.x] += red[threadIdx.x + st];
    __syncthreads();
  }
  float rstd = rsqrtf(red[0] * (1.0f / 2048.0f) + 1e-5f);
#pragma unroll
  for (int i = 0; i < 8; ++i) {
    int j = threadIdx.x * 8 + i;
    out[(size_t)row * 2048 + j] = (bh)((vals[i] - mean) * rstd * g[j] + bt[j]);
  }
}

// ---------------------------------------------------------------------------
// Generic GEMM: out[M][N] = A[M][K] @ W[N][K]^T + bias (+ exact GELU)
// ---------------------------------------------------------------------------
template <bool GELU>
__global__ __launch_bounds__(256) void gemm_wmma(
    const bh* __restrict__ A, const bh* __restrict__ Wt,
    const float* __restrict__ bias, bh* __restrict__ out,
    int M, int N, int K) {
  const int wave = threadIdx.x >> 5, lane = threadIdx.x & 31;
  const int ntiles = N >> 4, mtiles = M >> 4;
  const int tile = blockIdx.x * 8 + wave;
  if (tile >= ntiles * mtiles) return;
  const int mo = tile / ntiles, no = tile - mo * ntiles;
  const int hf = lane >> 4, l15 = lane & 15;
  const bh* arow = A + (size_t)(mo * 16 + l15) * K;
  const bh* brow = Wt + (size_t)(no * 16 + l15) * K;
  __builtin_prefetch(brow, 0, 0);
  float bi = bias[no * 16 + l15];
  v8f acc;
#pragma unroll
  for (int r = 0; r < 8; ++r) acc[r] = bi;
  for (int k0 = 0; k0 < K; k0 += 32)
    acc = wmma_bf16(fragA(arow, k0, hf), fragB(brow, k0, hf), acc);
#pragma unroll
  for (int r = 0; r < 8; ++r) {
    int m = mo * 16 + r + hf * 8;
    int n = no * 16 + l15;
    float v = acc[r];
    if (GELU) v = 0.5f * v * (1.0f + erff(v * 0.70710678118654752f));
    out[(size_t)m * N + n] = (bh)v;
  }
}

// ---------------------------------------------------------------------------
// Persistent single-workgroup LSTM (2 layers x 32 steps), WMMA gate GEMMs,
// h/c/gates resident in LDS (~176 KB of the WGP's 320 KB).
// ---------------------------------------------------------------------------
struct LayerP {
  const bh *wih, *whh, *resw, *memw;       // bf16 [N][K] row-major
  const float *bih, *bhh, *resb, *memb;
  const float *ilng, *ilnb, *hlng, *hlnb;
};

__global__ __launch_bounds__(256) void lstm_seq_k(
    const bh* __restrict__ proj,  // [512][256], row = b*32 + t
    LayerP L0, LayerP L1, float* __restrict__ hout /* [16][256] */) {
  __shared__ bh    s_xtn[16][256];
  __shared__ bh    s_hn[16][256];
  __shared__ float s_h[2][16][256];
  __shared__ float s_c[2][16][256];
  __shared__ float s_gates[16][1024];
  __shared__ float s_resm[16][512];

  const int tid = threadIdx.x, wave = tid >> 5, lane = tid & 31;
  const int hf = lane >> 4, l15 = lane & 15;

  for (int i = tid; i < 2 * 16 * 256; i += 256) {
    (&s_h[0][0][0])[i] = 0.0f;
    (&s_c[0][0][0])[i] = 0.0f;
  }
  __syncthreads();

  for (int t = 0; t < 32; ++t) {
    for (int layer = 0; layer < 2; ++layer) {
      const LayerP L = layer ? L1 : L0;

      // ---- LayerNorms (x and h), one wave per batch row, 2 passes of 8 rows
      for (int rp = 0; rp < 2; ++rp) {
        int row = rp * 8 + wave;
        float xv[8]; float s = 0.0f;
#pragma unroll
        for (int i = 0; i < 8; ++i) {
          int j = lane * 8 + i;
          float v = (layer == 0)
                        ? (float)proj[((size_t)row * 32 + t) * 256 + j]
                        : s_h[0][row][j];
          xv[i] = v; s += v;
        }
#pragma unroll
        for (int off = 16; off; off >>= 1) s += __shfl_xor(s, off, 32);
        float mu = s * (1.0f / 256.0f);
        float q = 0.0f;
#pragma unroll
        for (int i = 0; i < 8; ++i) { float d = xv[i] - mu; q += d * d; }
#pragma unroll
        for (int off = 16; off; off >>= 1) q += __shfl_xor(q, off, 32);
        float rs = rsqrtf(q * (1.0f / 256.0f) + 1e-5f);
#pragma unroll
        for (int i = 0; i < 8; ++i) {
          int j = lane * 8 + i;
          s_xtn[row][j] = (bh)((xv[i] - mu) * rs * L.ilng[j] + L.ilnb[j]);
        }
        float hv[8]; s = 0.0f;
#pragma unroll
        for (int i = 0; i < 8; ++i) {
          int j = lane * 8 + i;
          float v = s_h[layer][row][j];
          hv[i] = v; s += v;
        }
#pragma unroll
        for (int off = 16; off; off >>= 1) s += __shfl_xor(s, off, 32);
        mu = s * (1.0f / 256.0f);
        q = 0.0f;
#pragma unroll
        for (int i = 0; i < 8; ++i) { float d = hv[i] - mu; q += d * d; }
#pragma unroll
        for (int off = 16; off; off >>= 1) q += __shfl_xor(q, off, 32);
        rs = rsqrtf(q * (1.0f / 256.0f) + 1e-5f);
#pragma unroll
        for (int i = 0; i < 8; ++i) {
          int j = lane * 8 + i;
          s_hn[row][j] = (bh)((hv[i] - mu) * rs * L.hlng[j] + L.hlnb[j]);
        }
      }
      __syncthreads();

      // ---- gates = x_tn @ Wih^T + h_n @ Whh^T + b   (16 x 1024, K=256+256)
      for (int nt = wave; nt < 64; nt += 8) {
        int n = nt * 16 + l15;
        float bi = L.bih[n] + L.bhh[n];
        v8f acc;
#pragma unroll
        for (int r = 0; r < 8; ++r) acc[r] = bi;
        const bh* w1 = L.wih + (size_t)n * 256;
        const bh* w2 = L.whh + (size_t)n * 256;
        for (int k0 = 0; k0 < 256; k0 += 32)
          acc = wmma_bf16(fragA(&s_xtn[l15][0], k0, hf), fragB(w1, k0, hf), acc);
        for (int k0 = 0; k0 < 256; k0 += 32)
          acc = wmma_bf16(fragA(&s_hn[l15][0], k0, hf), fragB(w2, k0, hf), acc);
#pragma unroll
        for (int r = 0; r < 8; ++r)
          s_gates[r + hf * 8][nt * 16 + l15] = acc[r];
      }

      // ---- res / mem gates: [x_tn, h_n] @ W^T   (16 x 256 each, K=512)
      for (int nt = wave; nt < 32; nt += 8) {
        bool memg = nt >= 16;
        int ntl = memg ? nt - 16 : nt;
        int n = ntl * 16 + l15;
        const bh* wr = (memg ? L.memw : L.resw) + (size_t)n * 512;
        float bi = memg ? L.memb[n] : L.resb[n];
        v8f acc;
#pragma unroll
        for (int r = 0; r < 8; ++r) acc[r] = bi;
        for (int k0 = 0; k0 < 256; k0 += 32)
          acc = wmma_bf16(fragA(&s_xtn[l15][0], k0, hf), fragB(wr, k0, hf), acc);
        for (int k0 = 0; k0 < 256; k0 += 32)
          acc = wmma_bf16(fragA(&s_hn[l15][0], k0, hf), fragB(wr + 256, k0, hf), acc);
#pragma unroll
        for (int r = 0; r < 8; ++r)
          s_resm[r + hf * 8][(memg ? 256 : 0) + ntl * 16 + l15] = acc[r];
      }
      __syncthreads();

      // ---- elementwise cell update (torch gate order: i,f,g,o)
      for (int i = tid; i < 16 * 256; i += 256) {
        int m = i >> 8, j = i & 255;
        float gi = s_gates[m][j];
        float gf = s_gates[m][256 + j];
        float gg = s_gates[m][512 + j];
        float go = s_gates[m][768 + j];
        float c  = s_c[layer][m][j];
        float si = sigm(gi), sf = sigm(gf), so = sigm(go), tg = tanhf(gg);
        float cn = sf * c + si * tg;
        float hn_ = so * tanhf(cn);
        float gr = sigm(s_resm[m][j]);
        float h_out = gr * hn_ + (1.0f - gr) * s_h[layer][m][j];
        float gm = sigm(s_resm[m][256 + j]);
        float c_out = gm * cn + (1.0f - gm) * tanhf(cn);
        s_h[layer][m][j] = h_out;
        s_c[layer][m][j] = c_out;
      }
      __syncthreads();
    }
  }
  for (int i = tid; i < 16 * 256; i += 256)
    hout[i] = s_h[1][i >> 8][i & 255];
}

// ---------------------------------------------------------------------------
// Head: LN(256) then 6-class linear, f32 output. One block per batch row.
// ---------------------------------------------------------------------------
__global__ __launch_bounds__(256) void head_k(
    const float* __restrict__ hf, const float* __restrict__ g,
    const float* __restrict__ bt, const float* __restrict__ w,
    const float* __restrict__ bias, float* __restrict__ out) {
  const int row = blockIdx.x;
  __shared__ float red[256];
  __shared__ float xn[256];
  float v = hf[(size_t)row * 256 + threadIdx.x];
  red[threadIdx.x] = v; __syncthreads();
  for (int st = 128; st > 0; st >>= 1) {
    if (threadIdx.x < st) red[threadIdx.x] += red[threadIdx.x + st];
    __syncthreads();
  }
  float mean = red[0] * (1.0f / 256.0f);
  __syncthreads();
  float d = v - mean;
  red[threadIdx.x] = d * d; __syncthreads();
  for (int st = 128; st > 0; st >>= 1) {
    if (threadIdx.x < st) red[threadIdx.x] += red[threadIdx.x + st];
    __syncthreads();
  }
  float rstd = rsqrtf(red[0] * (1.0f / 256.0f) + 1e-5f);
  xn[threadIdx.x] = d * rstd * g[threadIdx.x] + bt[threadIdx.x];
  __syncthreads();
  if (threadIdx.x < 6) {
    float s = bias[threadIdx.x];
    for (int j = 0; j < 256; ++j) s += xn[j] * w[threadIdx.x * 256 + j];
    out[row * 6 + threadIdx.x] = s;
  }
}

// ---------------------------------------------------------------------------
// Host orchestration
// ---------------------------------------------------------------------------
// Input flattening (jax tree_leaves: dict keys sorted alphabetically per level)
enum {
  I_HEAD_B = 0, I_HEAD_LNB, I_HEAD_LNG, I_HEAD_W,
  I_RNN0 = 4,   // 12 leaves per layer
  I_SEQ_B = 28, I_SEQ_LNB, I_SEQ_LNG, I_SEQ_W,
  I_STAGE0 = 32, // 14 leaves per stage
  I_X = 88
};
enum { O_BHH = 0, O_BIH, O_HLNB, O_HLNG, O_ILNB, O_ILNG, O_MEMB, O_MEMW, O_RESB, O_RESW, O_WHH, O_WIH };
enum { O_BN1B = 0, O_BN1G, O_BN1M, O_BN1V, O_BN2B, O_BN2G, O_BN2M, O_BN2V,
       O_C1B, O_C1W, O_C2B, O_C2W, O_SKB, O_SKW };

struct StageCfg { int Ci, Co, H, W, NPW; };
static const StageCfg SC[4] = {{1, 32, 128, 512, 4}, {32, 64, 64, 256, 2},
                               {64, 128, 32, 128, 1}, {128, 256, 16, 64, 1}};

extern "C" void kernel_launch(void* const* d_in, const int* in_sizes, int n_in,
                              void* d_out, int out_size, void* d_ws, size_t ws_size,
                              hipStream_t stream) {
  (void)in_sizes; (void)n_in; (void)out_size; (void)ws_size;
  auto F = [&](int i) -> const float* { return (const float*)d_in[i]; };

  size_t off = 0;
  auto alloc = [&](size_t bytes) -> char* {
    char* p = (char*)d_ws + off;
    off = (off + bytes + 255) & ~(size_t)255;
    return p;
  };
  auto cvt = [&](const float* src, bh* dst, int n) {
    cvt_bf16_k<<<(n + 255) / 256, 256, 0, stream>>>(src, dst, n);
  };

  // ---- workspace carve ----
  bh* xbf = (bh*)alloc((size_t)16 * 128 * 512 * 2);
  bh* t1  = (bh*)alloc((size_t)33554432 * 2);
  bh* t2  = (bh*)alloc((size_t)33554432 * 2);
  bh* p0  = (bh*)alloc((size_t)8388608 * 2);
  bh* p1  = (bh*)alloc((size_t)4194304 * 2);
  bh* seqbuf  = (bh*)alloc((size_t)512 * 2048 * 2);
  bh* projbuf = (bh*)alloc((size_t)512 * 256 * 2);
  float* hfin = (float*)alloc((size_t)16 * 256 * 4);

  bh *w1b[4], *w2b[4], *skwb[4];
  float *sc1[4], *bi1[4], *sc2[4], *bi2[4];
  for (int s = 0; s < 4; ++s) {
    int base = I_STAGE0 + s * 14;
    int Ci = SC[s].Ci, Co = SC[s].Co;
    w1b[s]  = (bh*)alloc((size_t)Co * Ci * 9 * 2);
    w2b[s]  = (bh*)alloc((size_t)Co * Co * 9 * 2);
    skwb[s] = (bh*)alloc((size_t)Co * Ci * 2);
    sc1[s] = (float*)alloc(Co * 4); bi1[s] = (float*)alloc(Co * 4);
    sc2[s] = (float*)alloc(Co * 4); bi2[s] = (float*)alloc(Co * 4);
    cvt(F(base + O_C1W), w1b[s], Co * Ci * 9);
    cvt(F(base + O_C2W), w2b[s], Co * Co * 9);
    cvt(F(base + O_SKW), skwb[s], Co * Ci);
    fold_bn_k<<<(Co + 255) / 256, 256, 0, stream>>>(
        F(base + O_BN1G), F(base + O_BN1B), F(base + O_BN1M), F(base + O_BN1V),
        F(base + O_C1B), sc1[s], bi1[s], Co);
    fold_bn_k<<<(Co + 255) / 256, 256, 0, stream>>>(
        F(base + O_BN2G), F(base + O_BN2B), F(base + O_BN2M), F(base + O_BN2V),
        F(base + O_C2B), sc2[s], bi2[s], Co);
  }
  bh* seqw = (bh*)alloc((size_t)256 * 2048 * 2);
  cvt(F(I_SEQ_W), seqw, 256 * 2048);

  LayerP LP[2];
  for (int l = 0; l < 2; ++l) {
    int base = I_RNN0 + l * 12;
    bh* wih  = (bh*)alloc((size_t)1024 * 256 * 2);
    bh* whh  = (bh*)alloc((size_t)1024 * 256 * 2);
    bh* resw = (bh*)alloc((size_t)256 * 512 * 2);
    bh* memw = (bh*)alloc((size_t)256 * 512 * 2);
    cvt(F(base + O_WIH), wih, 1024 * 256);
    cvt(F(base + O_WHH), whh, 1024 * 256);
    cvt(F(base + O_RESW), resw, 256 * 512);
    cvt(F(base + O_MEMW), memw, 256 * 512);
    LP[l] = {wih, whh, resw, memw,
             F(base + O_BIH), F(base + O_BHH), F(base + O_RESB), F(base + O_MEMB),
             F(base + O_ILNG), F(base + O_ILNB), F(base + O_HLNG), F(base + O_HLNB)};
  }

  // ---- input to bf16 ----
  cvt(F(I_X), xbf, 16 * 128 * 512);

  // ---- CNN stages ----
  bh* stage_out[4] = {p0, p1, p0, p1};
  const bh* cur = xbf;
  for (int s = 0; s < 4; ++s) {
    int Ci = SC[s].Ci, Co = SC[s].Co, H = SC[s].H, W = SC[s].W;
    int NPW = SC[s].NPW, NPC = 8 / NPW;
    int SW = NPW * 16;
    int blocks = 16 * H * (W / SW);
    if (s < 3) {
      conv_tile_wmma<1, true><<<blocks, 256, 0, stream>>>(
          cur, w1b[s], sc1[s], bi1[s], t1, Ci, H, W, Co, SW, NPW, NPC);
      conv_tile_wmma<1, false><<<blocks, 256, 0, stream>>>(
          t1, w2b[s], sc2[s], bi2[s], t2, Co, H, W, Co, SW, NPW, NPC);
    } else {
      conv_tile_wmma<2, true><<<blocks, 256, 0, stream>>>(
          cur, w1b[s], sc1[s], bi1[s], t1, Ci, H, W, Co, SW, NPW, NPC);
      conv_tile_wmma<2, false><<<blocks, 256, 0, stream>>>(
          t1, w2b[s], sc2[s], bi2[s], t2, Co, H, W, Co, SW, NPW, NPC);
    }
    int base = I_STAGE0 + s * 14;
    int tot = 16 * Co * (H / 2) * (W / 2);
    skip_pool_k<<<(tot + 255) / 256, 256, 0, stream>>>(
        t2, cur, skwb[s], F(base + O_SKB), stage_out[s], 16, Ci, Co, H, W);
    cur = stage_out[s];
  }

  // ---- sequence projection: LN + GEMM(512x256x2048) + GELU ----
  ln_feat_k<<<512, 256, 0, stream>>>(cur, F(I_SEQ_LNG), F(I_SEQ_LNB), seqbuf);
  {
    int tiles = (512 / 16) * (256 / 16);
    gemm_wmma<true><<<(tiles + 7) / 8, 256, 0, stream>>>(
        seqbuf, seqw, F(I_SEQ_B), projbuf, 512, 256, 2048);
  }

  // ---- persistent LSTM ----
  lstm_seq_k<<<1, 256, 0, stream>>>(projbuf, LP[0], LP[1], hfin);

  // ---- head ----
  head_k<<<16, 256, 0, stream>>>(hfin, F(I_HEAD_LNG), F(I_HEAD_LNB),
                                 F(I_HEAD_W), F(I_HEAD_B), (float*)d_out);
}
